// BigResNet_65850438582498
// MI455X (gfx1250) — compile-verified, matrix operand order checked by reading, not run on hardware
//
#include <hip/hip_runtime.h>

// ---------------- problem constants ----------------
#define D_IN      100
#define NLAYERS   1000      // 100 blocks x 10 layers
#define NT        8         // N tiles of 16  (100 -> 128 padded)
#define KT        4         // K tiles of 32  (100 -> 128 padded)
#define ROWS      128       // batch rows per workgroup
#define STRIDE    136       // LDS activation row stride in bf16 (bank-conflict pad)
#define WGTHREADS 512       // 16 waves
#define ACT_ELEMS (ROWS*STRIDE)                // 17408 = 34 * 512
#define RES_PER_T (ACT_ELEMS/WGTHREADS)        // 34 residual elems per thread
#define FRAG_HALVES_PER_LAYER (NT*KT*32*16)    // 16384 bf16 = 32 KB

typedef __attribute__((ext_vector_type(16))) __bf16 v16bf;
typedef __attribute__((ext_vector_type(8)))  __bf16 v8bf;
typedef __attribute__((ext_vector_type(2)))  __bf16 v2bf;
typedef __attribute__((ext_vector_type(8)))  float  v8f;
typedef __attribute__((ext_vector_type(2)))  float  v2f;

__device__ __forceinline__ unsigned short f2bf(float f) {    // RNE, cold paths
    union { float f; unsigned int u; } v; v.f = f;
    unsigned int r = v.u + 0x7FFFu + ((v.u >> 16) & 1u);
    return (unsigned short)(r >> 16);
}
__device__ __forceinline__ float bf2f(unsigned short h) {
    union { unsigned int u; float f; } v; v.u = ((unsigned int)h) << 16;
    return v.f;
}
// hot path: packed f32x2 -> bf16x2 (lets backend pick v_cvt_pk_bf16_f32)
__device__ __forceinline__ void f2bf_pk(float a, float b,
                                        unsigned short* s0, unsigned short* s1) {
    v2f vv = { a, b };
    union { v2bf v; unsigned short s[2]; } u;
    u.v = __builtin_convertvector(vv, v2bf);
    *s0 = u.s[0]; *s1 = u.s[1];
}

// ---------------------------------------------------------------------------
// Pre-pass: pack W[1000][100][100] f32 -> bf16 B-fragments in WMMA lane order.
// Fragment layout: [layer][nt][kt][lane(32)][half(16)]
//   lane<16 : N = nt*16+lane,     K halves = {0..7, 16..23}  (+kt*32)
//   lane>=16: N = nt*16+lane-16,  K halves = {8..15, 24..31} (+kt*32)
// ---------------------------------------------------------------------------
__global__ void pack_weights_kernel(const float* __restrict__ W,
                                    unsigned short* __restrict__ out) {
    int id = blockIdx.x * 256 + threadIdx.x;     // one bf16 half per thread
    int h     = id & 15;
    int lane  = (id >> 4) & 31;
    int kt    = (id >> 9) & 3;
    int nt    = (id >> 11) & 7;
    int layer = id >> 14;
    if (layer >= NLAYERS) return;
    int n  = nt * 16 + (lane & 15);
    int hi = lane >> 4;
    int kl = (h < 8) ? (h + hi * 8) : ((h - 8) + 16 + hi * 8);
    int k  = kt * 32 + kl;
    float v = 0.0f;
    if (n < D_IN && k < D_IN)
        v = W[(size_t)layer * D_IN * D_IN + n * D_IN + k];
    out[id] = f2bf(v);
}

// ---------------------------------------------------------------------------
// Async weight staging: each thread issues 4 x b128 (2048 chunks total) from
// global (L2-resident packed weights) directly into LDS, tracked by ASYNCcnt.
// ---------------------------------------------------------------------------
__device__ __forceinline__ void stage_layer_async(const unsigned short* gsrc,
                                                  unsigned lds_base, int tid) {
    #pragma unroll
    for (int q = 0; q < 4; ++q) {
        unsigned byteoff = (unsigned)(tid + q * WGTHREADS) * 16u;
        asm volatile("global_load_async_to_lds_b128 %0, %1, %2"
                     :: "v"(lds_base + byteoff), "v"(byteoff), "s"(gsrc)
                     : "memory");
    }
}
__device__ __forceinline__ void wait_async0() {
    asm volatile("s_wait_asynccnt 0x0" ::: "memory");
}

// ---------------------------------------------------------------------------
// Main kernel: 512 WGs x 512 threads; each WG owns 128 batch rows, runs all
// 1000 layers with LDS ping-pong activations, double-buffered async-staged
// weights, and the block residual held in per-thread registers.
// ---------------------------------------------------------------------------
__global__ __launch_bounds__(WGTHREADS)
void bigresnet_wmma_kernel(const float* __restrict__ x,
                           const unsigned short* __restrict__ wpk,
                           const float* __restrict__ bias,
                           const float* __restrict__ Wf,
                           const float* __restrict__ bfv,
                           float* __restrict__ out) {
    __shared__ unsigned short actA [ACT_ELEMS];
    __shared__ unsigned short actB [ACT_ELEMS];
    __shared__ unsigned short wfrag[2 * FRAG_HALVES_PER_LAYER];   // 64 KB dbl-buf

    const int tid  = threadIdx.x;
    const int wave = tid >> 5;
    const int lane = tid & 31;
    const int r0   = blockIdx.x * ROWS;

    // residual x_b, strided across threads: element i = tid + j*512
    unsigned short rres[RES_PER_T];

    // ---- load x tile: f32 -> bf16, pad cols 100..135 with zero ----
    #pragma unroll
    for (int j = 0; j < RES_PER_T; ++j) {
        int i = tid + j * WGTHREADS;
        int m = i / STRIDE, c = i - m * STRIDE;
        unsigned short v = 0;
        if (c < D_IN) v = f2bf(x[(size_t)(r0 + m) * D_IN + c]);
        actA[i] = v; rres[j] = v;
    }
    // kick off async stage of layer 0 into buffer 0
    const unsigned wbase = (unsigned)(uintptr_t)&wfrag[0];
    stage_layer_async(wpk, wbase, tid);
    __syncthreads();

    const int nt   = wave & 7;        // this wave's N tile
    const int mt0  = wave >> 3;       // 0/1; handles mt = mt0, mt0+2, mt0+4, mt0+6
    const int hi   = lane >> 4;       // lane group (K-half select)
    const int ncol = nt * 16 + (lane & 15);

    unsigned short* src = actA;
    unsigned short* dst = actB;

    #pragma unroll 1
    for (int blk = 0; blk < 100; ++blk) {
        #pragma unroll 1
        for (int ly = 0; ly < 10; ++ly) {
            const int lidx = blk * 10 + ly;
            const int buf  = lidx & 1;

            // layer L's weights must have landed; make visible to all waves
            wait_async0();
            __syncthreads();
            // overlap: issue async stage of layer L+1 into the other buffer
            if (lidx + 1 < NLAYERS)
                stage_layer_async(wpk + (size_t)(lidx + 1) * FRAG_HALVES_PER_LAYER,
                                  wbase + (buf ? 0u : FRAG_HALVES_PER_LAYER * 2u), tid);

            const unsigned short* wcur = &wfrag[buf * FRAG_HALVES_PER_LAYER];
            const float bn = (ncol < D_IN) ? bias[lidx * D_IN + ncol] : 0.0f;

            // ---- B fragments: load once, reuse across 4 M tiles ----
            v16bf bfr[KT];
            #pragma unroll
            for (int kt = 0; kt < KT; ++kt)
                bfr[kt] = *(const v16bf*)&wcur[((nt * KT + kt) * 32 + lane) * 16];

            #pragma unroll
            for (int t = 0; t < 4; ++t) {
                const int mt  = mt0 + t * 2;
                const int row = mt * 16 + (lane & 15);
                v8f c = {};
                #pragma unroll
                for (int kt = 0; kt < KT; ++kt) {
                    const unsigned short* p = &src[row * STRIDE + kt * 32 + hi * 8];
                    union { v16bf v; v8bf h[2]; } a;
                    a.h[0] = *(const v8bf*)p;          // K {0..7}  (+hi*8)
                    a.h[1] = *(const v8bf*)(p + 16);   // K {16..23}(+hi*8)
                    c = __builtin_amdgcn_wmma_f32_16x16x32_bf16(
                            false, a.v, false, bfr[kt], (short)0, c, false, false);
                }
                // ---- fused bias + ReLU + packed bf16 convert ----
                const int mbase = mt * 16 + hi * 8;
                #pragma unroll
                for (int r = 0; r < 8; r += 2) {
                    float v0 = fmaxf(c[r]     + bn, 0.0f);
                    float v1 = fmaxf(c[r + 1] + bn, 0.0f);
                    f2bf_pk(v0, v1,
                            &dst[(mbase + r)     * STRIDE + ncol],
                            &dst[(mbase + r + 1) * STRIDE + ncol]);
                }
            }
            __syncthreads();
            unsigned short* tp = src; src = dst; dst = tp;   // ping-pong
        }
        // ---- residual: x_{b+1} = x_b + y  (registers + LDS, padding stays 0) ----
        #pragma unroll
        for (int j = 0; j < RES_PER_T; j += 2) {
            int i0 = tid + j * WGTHREADS;
            int i1 = i0 + WGTHREADS;
            float v0 = bf2f(rres[j])     + bf2f(src[i0]);
            float v1 = bf2f(rres[j + 1]) + bf2f(src[i1]);
            unsigned short h0, h1;
            f2bf_pk(v0, v1, &h0, &h1);
            rres[j] = h0; rres[j + 1] = h1;
            src[i0] = h0; src[i1] = h1;
        }
        __syncthreads();
    }

    // ---- final Linear 100 -> 10 (negligible FLOPs, scalar) ----
    for (int i = tid; i < ROWS * 10; i += WGTHREADS) {
        int m = i / 10, j = i - m * 10;
        float s = bfv[j];
        const unsigned short* ar = &src[m * STRIDE];
        const float*          wr = &Wf[j * D_IN];
        #pragma unroll 4
        for (int k = 0; k < D_IN; ++k) s += bf2f(ar[k]) * wr[k];
        out[(size_t)(r0 + m) * 10 + j] = s;
    }
}

// ---------------------------------------------------------------------------
extern "C" void kernel_launch(void* const* d_in, const int* in_sizes, int n_in,
                              void* d_out, int out_size, void* d_ws, size_t ws_size,
                              hipStream_t stream) {
    const float* x  = (const float*)d_in[0];   // [65536,100]
    const float* W  = (const float*)d_in[1];   // [100,10,100,100]
    const float* b  = (const float*)d_in[2];   // [100,10,100]
    const float* Wf = (const float*)d_in[3];   // [10,100]
    const float* bf = (const float*)d_in[4];   // [10]
    unsigned short* wpk = (unsigned short*)d_ws;  // 32.8 MB packed bf16 weights

    const int total_halves = NLAYERS * FRAG_HALVES_PER_LAYER;  // 1000 << 14
    pack_weights_kernel<<<total_halves / 256, 256, 0, stream>>>(W, wpk);

    bigresnet_wmma_kernel<<<65536 / ROWS, WGTHREADS, 0, stream>>>(
        x, wpk, b, Wf, bf, (float*)d_out);
}
